// DNNModel_61984968016173
// MI455X (gfx1250) — matrix operand order repaired and checked by reading, not compiled, not run on hardware
//
#include <hip/hip_runtime.h>

// ---------------------------------------------------------------------------
// Fused 9-layer MLP (Linear->ReLU->Dropout x8, Linear) for MI455X (gfx1250).
//
// Roofline: ~48.7 GFLOP vs ~432 MB mandatory HBM traffic (x + 8 dropout masks
// + out) => ~18.5us memory floor @ 23.3 TB/s. f16 WMMA (16x16x32, f32 accum)
// puts compute at ~the same level; fp32 WMMA (K=4) would be 8x compute-bound.
// => single fused kernel, activations ping-pong in LDS as f16, weights
// pre-packed once (fp32->f16, WMMA B-fragment layout) into d_ws; weights are
// ~1.5MB total so all fragment reads hit the 192MB L2.
// ---------------------------------------------------------------------------

typedef __attribute__((ext_vector_type(16))) _Float16 v16h;
typedef __attribute__((ext_vector_type(8)))  _Float16 v8h;
typedef __attribute__((ext_vector_type(8)))  float    v8f;

#define BATCH    65536
#define MT       64            // batch rows per workgroup
#define NWAVES   8
#define NTHREADS (NWAVES * 32) // wave32: 8 waves

// DIMS = 256,128,256,512,256,128,64,32,16,10
// Fragment counts per layer: (ceil32(K)/32) * (ceil16(N)/16)
//  L0:64 L1:64 L2:256 L3:256 L4:64 L5:16 L6:4 L7:1 L8:1  (total 726 * 1KB)

// --------------------------- weight pre-pack -------------------------------
// Pack W (fp32, KxN row-major) into WMMA B-operand fragments (32x16 f16).
// B layout (ISA 7.12.2, 16-bit B 32x16): lanes 0-15: N=lane,   K=kb+hh+8*(hh>=8)
//                                        lanes16-31: N=lane-16, K=kb+8+hh+8*(hh>=8)
// One block = one fragment; lane writes its 16 contiguous halves.
__global__ void prep_pack(const float* __restrict__ W, _Float16* __restrict__ pack,
                          int Kreal, int Nreal, int Kpad) {
    const int frag    = blockIdx.x;
    const int kBlocks = Kpad >> 5;
    const int kb      = (frag % kBlocks) * 32;
    const int nb      = (frag / kBlocks) * 16;
    const int lane    = threadIdx.x & 31;
    const int n       = nb + (lane & 15);
    const int k0      = kb + ((lane >> 4) << 3);
    _Float16* dst = pack + (size_t)frag * 512 + lane * 16;
#pragma unroll
    for (int hh = 0; hh < 16; ++hh) {
        const int k = k0 + hh + ((hh >> 3) << 3);
        float v = 0.0f;
        if (k < Kreal && n < Nreal) v = W[(size_t)k * Nreal + n];
        dst[hh] = (_Float16)v;   // zero-pad K>=Kreal (layer9 K=16) and N>=Nreal (final N=10)
    }
}

// ----------------------------- fused MLP -----------------------------------
struct MlpArgs {
    const float*    x;
    const _Float16* pack;
    float*          out;
    const float*    b[9];
    const float*    m[8];
};

__global__ __launch_bounds__(NTHREADS) void mlp_fused(MlpArgs p) {
    constexpr int DIMS[10]    = {256, 128, 256, 512, 256, 128, 64, 32, 16, 10};
    constexpr int FRAGBASE[9] = {0, 64, 128, 384, 640, 704, 720, 724, 725};

    extern __shared__ __attribute__((aligned(16))) _Float16 smem[];
    _Float16* bufs[2];
    bufs[0] = smem;             // even-layer inputs: max stride 256+8 = 264 halves
    bufs[1] = smem + MT * 264;  // odd-layer inputs:  max stride 512+8 = 520 halves

    const int    tid     = threadIdx.x;
    const int    lane    = tid & 31;
    const int    wave    = tid >> 5;
    const int    lane15  = lane & 15;
    const int    hiHalf  = lane >> 4;           // 0 for lanes 0-15, 1 for 16-31
    const size_t rowBase = (size_t)blockIdx.x * MT;

    // Stage x tile (64 x 256 fp32 -> f16 in LDS, stride 264). Coalesced reads.
    for (int i = tid; i < MT * 256; i += NTHREADS) {
        const int r = i >> 8;
        const int c = i & 255;
        bufs[0][r * 264 + c] = (_Float16)p.x[(rowBase + r) * 256 + c];
    }
    __syncthreads();

#pragma unroll
    for (int l = 0; l < 9; ++l) {
        const int Kreal   = DIMS[l];
        const int Kpad    = (Kreal + 31) & ~31;
        const int Nreal   = DIMS[l + 1];
        const int Npad16  = (Nreal + 15) & ~15;
        const int NpadOut = (Nreal + 31) & ~31;   // out buffer = next layer's Kpad
        const int sIn     = Kpad + 8;             // +8 halves: LDS bank-conflict pad
        const int sOut    = NpadOut + 8;
        const int kBlocks = Kpad >> 5;
        const _Float16* in   = bufs[l & 1];
        _Float16*       outb = bufs[(l & 1) ^ 1];
        const int  nTiles = (MT / 16) * (Npad16 / 16);
        const bool last   = (l == 8);

        for (int t = wave; t < nTiles; t += NWAVES) {
            const int rb = (t & (MT / 16 - 1)) * 16;
            const int nb = (t >> 2) * 16;
            v8f acc = {};
            // A layout (16-bit A 16x32): lanes 0-15 get K=[kb..kb+7],[kb+16..kb+23];
            // lanes 16-31 get K=[kb+8..kb+15],[kb+24..kb+31]. Two ds_load_b128 each.
            const _Float16* aRow  = in + (rb + lane15) * sIn + hiHalf * 8;
            const _Float16* bFrag = p.pack +
                ((size_t)FRAGBASE[l] + (size_t)(nb >> 4) * kBlocks) * 512 + lane * 16;
            for (int kb = 0; kb < Kpad; kb += 32) {
                v8h alo = *(const v8h*)(aRow + kb);
                v8h ahi = *(const v8h*)(aRow + kb + 16);
                v16h a  = __builtin_shufflevector(alo, ahi,
                              0, 1, 2, 3, 4, 5, 6, 7, 8, 9, 10, 11, 12, 13, 14, 15);
                v16h bm = *(const v16h*)(bFrag + (size_t)(kb >> 5) * 512);
                acc = __builtin_amdgcn_wmma_f32_16x16x32_f16(
                          false, a, false, bm, (short)0, acc, false, false);
            }
            // Epilogue. D layout: lanes 0-15: N=nb+lane, M=rb+v; lanes 16-31: M=rb+8+v.
            const int   n    = nb + lane15;
            const float bias = (n < Nreal) ? p.b[l][n] : 0.0f;
            if (!last) {
                const float* mp = p.m[l];
#pragma unroll
                for (int v = 0; v < 8; ++v) {
                    const int rloc = rb + hiHalf * 8 + v;
                    float val = acc[v] + bias;
                    val = val > 0.0f ? val : 0.0f;                    // ReLU
                    val *= mp[(rowBase + rloc) * (size_t)Nreal + n];  // inverted dropout
                    outb[rloc * sOut + n] = (_Float16)val;
                }
            } else {
#pragma unroll
                for (int v = 0; v < 8; ++v) {
                    const int rloc = rb + hiHalf * 8 + v;
                    if (n < Nreal)
                        p.out[(rowBase + rloc) * (size_t)Nreal + n] = acc[v] + bias;
                }
            }
        }

        // Zero-fill K padding columns for the next layer (only dim 16 -> pad 32).
        if (!last && NpadOut > Nreal) {
            const int padCols = NpadOut - Nreal;
            for (int i = tid; i < MT * padCols; i += NTHREADS) {
                const int r = i / padCols;
                const int c = Nreal + (i % padCols);
                outb[r * sOut + c] = (_Float16)0.0f;
            }
        }
        __syncthreads();  // covers reads of `in` vs next layer's writes (ping-pong)
    }
}

// ------------------------------- launcher ----------------------------------
extern "C" void kernel_launch(void* const* d_in, const int* in_sizes, int n_in,
                              void* d_out, int out_size, void* d_ws, size_t ws_size,
                              hipStream_t stream) {
    static const int DIMS[10]    = {256, 128, 256, 512, 256, 128, 64, 32, 16, 10};
    static const int FRAGBASE[9] = {0, 64, 128, 384, 640, 704, 720, 724, 725};

    // d_ws: 726 fragments * 512 f16 = 743,424 bytes of packed weights.
    _Float16* pack = (_Float16*)d_ws;

    // Re-pack weights every call (deterministic; ~0.7 MB, negligible).
    for (int l = 0; l < 9; ++l) {
        const int Kreal = DIMS[l], Nreal = DIMS[l + 1];
        const int Kpad  = (Kreal + 31) & ~31;
        const int Npad  = (Nreal + 15) & ~15;
        const int nFrag = (Kpad / 32) * (Npad / 16);
        const float* W  = (const float*)d_in[1 + 2 * l];
        prep_pack<<<nFrag, 32, 0, stream>>>(W, pack + (size_t)FRAGBASE[l] * 512,
                                            Kreal, Nreal, Kpad);
    }

    MlpArgs a;
    a.x    = (const float*)d_in[0];
    a.pack = pack;
    a.out  = (float*)d_out;
    for (int i = 0; i < 9; ++i) a.b[i] = (const float*)d_in[2 + 2 * i];
    for (int i = 0; i < 8; ++i) a.m[i] = (const float*)d_in[19 + i];

    const int smemBytes = (MT * 264 + MT * 520) * (int)sizeof(_Float16); // 100,352 B
    mlp_fused<<<BATCH / MT, NTHREADS, smemBytes, stream>>>(a);
}
